// LSTM_vowel_recognizer_20564303413410
// MI455X (gfx1250) — matrix operand order repaired and checked by reading, not compiled, not run on hardware
//
#include <hip/hip_runtime.h>

// ---------------------------------------------------------------------------
// LSTM vowel recognizer for MI455X (gfx1250), bf16 WMMA pipeline.
//   seq (scan) = 64, batch = 512, H = 256, gates = 1024, K0 = 1625 (pad 1664)
// XG gate-preactivation tensor is stored COLUMN-major (col*NROWS + row) so
// both the GEMM store side and the recurrence load side are b128 vector ops.
// ---------------------------------------------------------------------------

#define BT    64          // scan length (input B axis)
#define TT    512         // inner batch (input T axis)
#define NROWS (BT * TT)   // 32768
#define FIN   65
#define K0    1625
#define K0P   1664        // padded to multiple of 32
#define HID   256
#define GAT   1024

typedef __attribute__((ext_vector_type(16))) __bf16 v16bf;
typedef __attribute__((ext_vector_type(8)))  float  v8f;
union FragAB { uint4 q[2]; v16bf v; };

__device__ __forceinline__ unsigned short f2bf(float f) {
  unsigned int u = __float_as_uint(f);
  u += 0x7FFFu + ((u >> 16) & 1u);          // round-to-nearest-even
  return (unsigned short)(u >> 16);
}
__device__ __forceinline__ float sigm(float x)   { return 1.0f / (1.0f + __expf(-x)); }
__device__ __forceinline__ float tanhf_(float x) { return 1.0f - 2.0f / (__expf(2.0f * x) + 1.0f); }

// ---------------- packing kernels ----------------
__global__ void cvt_pad_k(unsigned short* __restrict__ dst, const float* __restrict__ src,
                          int rows, int Ks, int Kd) {
  int i = blockIdx.x * 256 + threadIdx.x;
  if (i >= rows * Kd) return;
  int r = i / Kd, k = i - r * Kd;
  dst[i] = (k < Ks) ? f2bf(src[r * Ks + k]) : (unsigned short)0;
}

__global__ void bias_k(float* __restrict__ dst, const float* __restrict__ a,
                       const float* __restrict__ b, int n) {
  int i = blockIdx.x * 256 + threadIdx.x;
  if (i < n) dst[i] = a[i] + b[i];
}

// Build framed bf16 A0: A0[(s*512+b)][j*65+f] = input[s, b-12+j, f] (zero padded)
__global__ void pack_a0_k(unsigned short* __restrict__ dst, const float* __restrict__ in) {
  int i = blockIdx.x * 256 + threadIdx.x;
  if (i >= NROWS * K0P) return;
  int row = i / K0P, k = i - row * K0P;
  int s = row >> 9, b = row & 511;
  float v = 0.0f;
  if (k < K0) {
    int j = k / FIN, f = k - j * FIN;
    int t = b - 12 + j;
    if (t >= 0 && t < TT) v = in[((s << 9) + t) * FIN + f];
  }
  dst[i] = f2bf(v);
}

// ---------------- XG = A(bf16, NROWSxK) * W(bf16, 1024xK)^T + bias -------------
// One wave computes a 32(M) x 64(N) tile: 2 A frags x 4 B frags -> 8 wmma/K-step,
// 12 b128 loads per K-step (1.5 loads per wmma). K stepped by 32.
// A frag (ISA 7.12.2): lane -> row M=lane%16, halves at k+(lane/16)*8 .. +7, +16..+23.
// B frag (K x N):      lane -> col N=lane%16, halves at k+(lane/16)*8 .. +7, +16..+23 (as A).
// Output written COLUMN-major: XG[col * NROWS + row].
__global__ __launch_bounds__(256) void gemm_xg_k(
    const unsigned short* __restrict__ A, const unsigned short* __restrict__ W,
    const float* __restrict__ bias, float* __restrict__ XG, int K) {
  const int wv   = (blockIdx.x * 256 + threadIdx.x) >> 5;
  const int lane = threadIdx.x & 31;
  const int mt   = wv >> 4;           // 32-row M tile
  const int ns   = wv & 15;           // 64-col N strip
  const int l16  = lane & 15, hl = lane >> 4;

  const unsigned short* Arow0 = A + (size_t)(mt * 32 + l16) * K + hl * 8;
  const unsigned short* Arow1 = Arow0 + (size_t)16 * K;
  const unsigned short* Brow[4];
#pragma unroll
  for (int t = 0; t < 4; ++t)
    Brow[t] = W + (size_t)(ns * 64 + t * 16 + l16) * K + hl * 8;

  const v8f vzero = {0.f, 0.f, 0.f, 0.f, 0.f, 0.f, 0.f, 0.f};
  v8f acc[2][4];
#pragma unroll
  for (int m = 0; m < 2; ++m)
#pragma unroll
    for (int t = 0; t < 4; ++t) acc[m][t] = vzero;

  for (int k = 0; k < K; k += 32) {
    FragAB a0, a1;
    a0.q[0] = *(const uint4*)(Arow0 + k);
    a0.q[1] = *(const uint4*)(Arow0 + k + 16);
    a1.q[0] = *(const uint4*)(Arow1 + k);
    a1.q[1] = *(const uint4*)(Arow1 + k + 16);
#pragma unroll
    for (int t = 0; t < 4; ++t) {
      FragAB b;
      b.q[0] = *(const uint4*)(Brow[t] + k);
      b.q[1] = *(const uint4*)(Brow[t] + k + 16);
      acc[0][t] = __builtin_amdgcn_wmma_f32_16x16x32_bf16(
          false, a0.v, false, b.v, (short)0, acc[0][t], false, false);
      acc[1][t] = __builtin_amdgcn_wmma_f32_16x16x32_bf16(
          false, a1.v, false, b.v, (short)0, acc[1][t], false, false);
    }
  }
  // D layout: element r -> (M = r + 8*(lane/16), N = lane%16); 8 consecutive M rows
  // per lane -> two b128 stores into column-major XG.
#pragma unroll
  for (int m = 0; m < 2; ++m)
#pragma unroll
    for (int t = 0; t < 4; ++t) {
      const int col = ns * 64 + t * 16 + l16;
      const float bv = bias[col];
      float* dst = XG + (size_t)col * NROWS + (mt * 32 + m * 16 + hl * 8);
      float4 lo, hi;
      lo.x = acc[m][t][0] + bv; lo.y = acc[m][t][1] + bv;
      lo.z = acc[m][t][2] + bv; lo.w = acc[m][t][3] + bv;
      hi.x = acc[m][t][4] + bv; hi.y = acc[m][t][5] + bv;
      hi.z = acc[m][t][6] + bv; hi.w = acc[m][t][7] + bv;
      *(float4*)(dst)     = lo;
      *(float4*)(dst + 4) = hi;
    }
}

// ---------------- recurrence: rows independent across b, 64 sequential steps ----
// Block = 16 b-rows, 8 waves; wave owns 32 channels x 4 gates (8 accumulators).
// h lives in LDS as bf16 (WMMA A operand); c lives in registers in D layout.
// Accumulators are initialized from column-major XG with b128 loads.
__global__ __launch_bounds__(256) void lstm_rec_k(
    const float* __restrict__ XG, const unsigned short* __restrict__ U,
    unsigned short* __restrict__ Hb, float* __restrict__ Hf) {
  __shared__ unsigned short hbuf[16][HID];
  const int bBase = blockIdx.x * 16;
  const int lane  = threadIdx.x & 31;
  const int wvid  = threadIdx.x >> 5;
  const int l16   = lane & 15, hl = lane >> 4;
  const int chBase = wvid * 32;

  for (int i = threadIdx.x; i < 16 * HID; i += 256) ((unsigned short*)hbuf)[i] = 0;

  const v8f vzero = {0.f, 0.f, 0.f, 0.f, 0.f, 0.f, 0.f, 0.f};
  v8f c[2];
  c[0] = vzero; c[1] = vzero;

  const unsigned short* Urow[4][2];
  const float* XGcol[4][2];
#pragma unroll
  for (int q = 0; q < 4; ++q)
#pragma unroll
    for (int t = 0; t < 2; ++t) {
      const int col = q * HID + chBase + t * 16 + l16;
      Urow[q][t]  = U + (size_t)col * HID + hl * 8;
      XGcol[q][t] = XG + (size_t)col * NROWS + bBase + hl * 8;
    }

  __syncthreads();

  for (int s = 0; s < BT; ++s) {
    v8f acc[4][2];
#pragma unroll
    for (int q = 0; q < 4; ++q)
#pragma unroll
      for (int t = 0; t < 2; ++t) {
        const float4 lo = *(const float4*)(XGcol[q][t] + s * TT);
        const float4 hi = *(const float4*)(XGcol[q][t] + s * TT + 4);
        acc[q][t][0] = lo.x; acc[q][t][1] = lo.y;
        acc[q][t][2] = lo.z; acc[q][t][3] = lo.w;
        acc[q][t][4] = hi.x; acc[q][t][5] = hi.y;
        acc[q][t][6] = hi.z; acc[q][t][7] = hi.w;
      }
#pragma unroll
    for (int kk = 0; kk < HID; kk += 32) {
      FragAB a;
      a.q[0] = *(const uint4*)&hbuf[l16][kk + hl * 8];
      a.q[1] = *(const uint4*)&hbuf[l16][kk + hl * 8 + 16];
#pragma unroll
      for (int q = 0; q < 4; ++q)
#pragma unroll
        for (int t = 0; t < 2; ++t) {
          FragAB b;
          b.q[0] = *(const uint4*)(Urow[q][t] + kk);
          b.q[1] = *(const uint4*)(Urow[q][t] + kk + 16);
          acc[q][t] = __builtin_amdgcn_wmma_f32_16x16x32_bf16(
              false, a.v, false, b.v, (short)0, acc[q][t], false, false);
        }
    }
    __syncthreads();   // all hbuf reads done before overwrite
#pragma unroll
    for (int t = 0; t < 2; ++t) {
      const int ch = chBase + t * 16 + l16;
#pragma unroll
      for (int r = 0; r < 8; ++r) {
        float iv = sigm(acc[0][t][r]);
        float fv = sigm(acc[1][t][r]);
        float gv = tanhf_(acc[2][t][r]);
        float ov = sigm(acc[3][t][r]);
        float cv = fv * c[t][r] + iv * gv;
        c[t][r] = cv;
        float hv = ov * tanhf_(cv);
        unsigned short hb = f2bf(hv);
        int m = hl * 8 + r;
        hbuf[m][ch] = hb;
        size_t o = (size_t)(s * TT + bBase + m) * HID + ch;
        Hb[o] = hb;
        Hf[o] = hv;
      }
    }
    __syncthreads();   // writes visible before next step's reads
  }
}

// ---------------- batch-norm stats + output head ----------------
__global__ __launch_bounds__(256) void bn_stage1_k(const float* __restrict__ Hf,
                                                   float* __restrict__ psum,
                                                   float* __restrict__ psq) {
  const int bl = blockIdx.x, t = threadIdx.x;     // 256 blocks x 256 ch
  float s = 0.f, q = 0.f;
  for (int r = bl * 128; r < bl * 128 + 128; ++r) {
    float v = Hf[(size_t)r * HID + t];
    s += v; q += v * v;
  }
  psum[bl * 256 + t] = s;
  psq [bl * 256 + t] = q;
}

__global__ void bn_stage2_k(const float* __restrict__ psum, const float* __restrict__ psq,
                            const float* __restrict__ gamma, const float* __restrict__ beta,
                            float* __restrict__ aArr, float* __restrict__ bArr) {
  const int t = threadIdx.x;
  float s = 0.f, q = 0.f;
  for (int bl = 0; bl < 256; ++bl) { s += psum[bl * 256 + t]; q += psq[bl * 256 + t]; }
  float mean = s * (1.0f / NROWS);
  float var  = q * (1.0f / NROWS) - mean * mean;
  float rstd = rsqrtf(var + 1e-5f);
  float a = rstd * gamma[t];
  aArr[t] = a;
  bArr[t] = beta[t] - mean * a;
}

__global__ __launch_bounds__(256) void out_head_k(const float* __restrict__ Hf,
                                                  const float* __restrict__ aArr,
                                                  const float* __restrict__ bArr,
                                                  const float* __restrict__ Wout,
                                                  const float* __restrict__ bout,
                                                  float* __restrict__ out) {
  const int row  = (blockIdx.x * 256 + threadIdx.x) >> 5;   // one wave per row
  const int lane = threadIdx.x & 31;
  if (row >= NROWS) return;
  float acc[6] = {0.f, 0.f, 0.f, 0.f, 0.f, 0.f};
  const float* h = Hf + (size_t)row * HID;
#pragma unroll
  for (int j = 0; j < 8; ++j) {
    int ch = lane * 8 + j;
    float x = h[ch] * aArr[ch] + bArr[ch];
    x = x > 0.f ? x : 0.f;
#pragma unroll
    for (int o = 0; o < 6; ++o) acc[o] += x * Wout[o * HID + ch];
  }
  for (int off = 16; off > 0; off >>= 1)
#pragma unroll
    for (int o = 0; o < 6; ++o) acc[o] += __shfl_down(acc[o], off, 32);
  if (lane == 0)
#pragma unroll
    for (int o = 0; o < 6; ++o) out[(size_t)row * 6 + o] = acc[o] + bout[o];
}

// ---------------------------------------------------------------------------
extern "C" void kernel_launch(void* const* d_in, const int* in_sizes, int n_in,
                              void* d_out, int out_size, void* d_ws, size_t ws_size,
                              hipStream_t stream) {
  (void)in_sizes; (void)n_in; (void)out_size; (void)ws_size;
  const float* in     = (const float*)d_in[0];
  const float* Wih[3] = {(const float*)d_in[1], (const float*)d_in[5], (const float*)d_in[9]};
  const float* Whh[3] = {(const float*)d_in[2], (const float*)d_in[6], (const float*)d_in[10]};
  const float* bih[3] = {(const float*)d_in[3], (const float*)d_in[7], (const float*)d_in[11]};
  const float* bhh[3] = {(const float*)d_in[4], (const float*)d_in[8], (const float*)d_in[12]};
  const float* gamma  = (const float*)d_in[13];
  const float* beta   = (const float*)d_in[14];
  const float* Wout   = (const float*)d_in[15];
  const float* bout   = (const float*)d_in[16];
  float* out = (float*)d_out;

  // workspace carve-out (256B aligned); total ~318 MB
  char* p = (char*)d_ws;
  auto take = [&](size_t bytes) -> char* {
    char* r = p; p += (bytes + 255) & ~(size_t)255; return r;
  };
  unsigned short* A0b = (unsigned short*)take((size_t)NROWS * K0P * 2);
  unsigned short* W0b = (unsigned short*)take((size_t)GAT * K0P * 2);
  unsigned short* W1b = (unsigned short*)take((size_t)GAT * HID * 2);
  unsigned short* W2b = (unsigned short*)take((size_t)GAT * HID * 2);
  unsigned short* U0b = (unsigned short*)take((size_t)GAT * HID * 2);
  unsigned short* U1b = (unsigned short*)take((size_t)GAT * HID * 2);
  unsigned short* U2b = (unsigned short*)take((size_t)GAT * HID * 2);
  float* bias0 = (float*)take(GAT * 4);
  float* bias1 = (float*)take(GAT * 4);
  float* bias2 = (float*)take(GAT * 4);
  float* XG    = (float*)take((size_t)NROWS * GAT * 4);
  unsigned short* Hb0 = (unsigned short*)take((size_t)NROWS * HID * 2);
  unsigned short* Hb1 = (unsigned short*)take((size_t)NROWS * HID * 2);
  float* Hf   = (float*)take((size_t)NROWS * HID * 4);
  float* psum = (float*)take(256 * 256 * 4);
  float* psq  = (float*)take(256 * 256 * 4);
  float* aArr = (float*)take(256 * 4);
  float* bArr = (float*)take(256 * 4);

  // ---- pack weights / biases / framed input ----
  cvt_pad_k<<<(GAT * K0P + 255) / 256, 256, 0, stream>>>(W0b, Wih[0], GAT, K0, K0P);
  cvt_pad_k<<<(GAT * HID + 255) / 256, 256, 0, stream>>>(W1b, Wih[1], GAT, HID, HID);
  cvt_pad_k<<<(GAT * HID + 255) / 256, 256, 0, stream>>>(W2b, Wih[2], GAT, HID, HID);
  cvt_pad_k<<<(GAT * HID + 255) / 256, 256, 0, stream>>>(U0b, Whh[0], GAT, HID, HID);
  cvt_pad_k<<<(GAT * HID + 255) / 256, 256, 0, stream>>>(U1b, Whh[1], GAT, HID, HID);
  cvt_pad_k<<<(GAT * HID + 255) / 256, 256, 0, stream>>>(U2b, Whh[2], GAT, HID, HID);
  bias_k<<<4, 256, 0, stream>>>(bias0, bih[0], bhh[0], GAT);
  bias_k<<<4, 256, 0, stream>>>(bias1, bih[1], bhh[1], GAT);
  bias_k<<<4, 256, 0, stream>>>(bias2, bih[2], bhh[2], GAT);
  pack_a0_k<<<(NROWS * (size_t)K0P + 255) / 256, 256, 0, stream>>>(A0b, in);

  const int gemmBlocks = (NROWS / 32) * (GAT / 64) / 8;   // 2048 (8 waves/block)

  // ---- layer 0 ----
  gemm_xg_k<<<gemmBlocks, 256, 0, stream>>>(A0b, W0b, bias0, XG, K0P);
  lstm_rec_k<<<TT / 16, 256, 0, stream>>>(XG, U0b, Hb0, Hf);
  // ---- layer 1 ----
  gemm_xg_k<<<gemmBlocks, 256, 0, stream>>>(Hb0, W1b, bias1, XG, HID);
  lstm_rec_k<<<TT / 16, 256, 0, stream>>>(XG, U1b, Hb1, Hf);
  // ---- layer 2 ----
  gemm_xg_k<<<gemmBlocks, 256, 0, stream>>>(Hb1, W2b, bias2, XG, HID);
  lstm_rec_k<<<TT / 16, 256, 0, stream>>>(XG, U2b, Hb0, Hf);

  // ---- batch norm + relu + output head ----
  bn_stage1_k<<<256, 256, 0, stream>>>(Hf, psum, psq);
  bn_stage2_k<<<1, 256, 0, stream>>>(psum, psq, gamma, beta, aArr, bArr);
  out_head_k<<<NROWS / 8, 256, 0, stream>>>(Hf, aArr, bArr, Wout, bout, out);
}
